// SOMFNN_25907242730006
// MI455X (gfx1250) — matrix operand order repaired and checked by reading, not compiled, not run on hardware
//
#include <hip/hip_runtime.h>
#include <math.h>

typedef float    v2f  __attribute__((ext_vector_type(2)));
typedef float    v4f  __attribute__((ext_vector_type(4)));
typedef float    v8f  __attribute__((ext_vector_type(8)));
typedef _Float16 v16h __attribute__((ext_vector_type(16)));

#define BN 1024
#define DN 64
#define ON 64
#define MN 1024
#define DELTA_V 0.13533528323661270f   // exp(-2)

// ---- workspace layout (float offsets into d_ws) ----
#define OFF_SEN    0                       // [1024] sum(x*x) per sample
#define OFF_GTERM  1024                    // [1]
#define OFF_NRULES 1025                    // [1] (int)
#define OFF_SENC   2048                    // [1024]
#define OFF_CNORM  3072                    // [1024]
#define OFF_SUP    4096                    // [1024] (int)
#define OFF_PN     5120                    // [1024] ||proto||^2
#define OFF_STAU   6144                    // [1024]
#define OFF_INVC   7168                    // [1024] 1/colsum
#define OFF_PROT   8192                    // [1024*64]
#define OFF_CENT   (OFF_PROT + MN*DN)      // [1024*64]
#define OFF_YN     (OFF_CENT + MN*DN)      // [1024*64]
#define OFF_DENS   (OFF_YN + BN*ON)        // [1024*1024]

// ---------------------------------------------------------------------------
// Kernel A: sen[b] = ||x_b||^2 ; g_term = mean(sen)/B - ||mean(x)/B||^2
// ---------------------------------------------------------------------------
__global__ __launch_bounds__(1024, 1)
void k_stats(const float* __restrict__ x, float* __restrict__ ws) {
    __shared__ float red[1024];
    __shared__ float gm2[DN];
    int tid = threadIdx.x;
    const float* xr = x + tid * DN;
    float s = 0.f;
    for (int d = 0; d < DN; ++d) { float v = xr[d]; s += v * v; }
    ws[OFF_SEN + tid] = s;
    red[tid] = s;
    __syncthreads();
    for (int off = 512; off > 0; off >>= 1) {
        if (tid < off) red[tid] += red[tid + off];
        __syncthreads();
    }
    float sensum = red[0];
    if (tid < DN) {
        float cs = 0.f;
        for (int b = 0; b < BN; ++b) cs += x[b * DN + tid];   // coalesced across lanes
        float gm = cs / ((float)BN * (float)BN);
        gm2[tid] = gm * gm;
    }
    __syncthreads();
    if (tid == 0) {
        float gsq = 0.f;
        for (int d = 0; d < DN; ++d) gsq += gm2[d];
        float gsm = sensum / ((float)BN * (float)BN);
        ws[OFF_GTERM] = gsm - gsq;
    }
}

// ---------------------------------------------------------------------------
// Kernel B: sequential rule scan. One workgroup, 1024 threads (32 waves).
// protos live in LDS (row stride 65 floats -> bank-conflict-free), senc/cnorm
// in LDS; cents/sup in global ws. ~283 KB dynamic LDS (CDNA5: 320 KB/WGP).
// ---------------------------------------------------------------------------
__global__ __launch_bounds__(1024, 1)
void k_scan(const float* __restrict__ x, float* __restrict__ ws) {
    extern __shared__ float smem[];
    float* protosL = smem;                   // 1024*65
    float* sencL   = protosL + MN * 65;      // 1024
    float* cnormL  = sencL + MN;             // 1024
    float* xiL     = cnormL + MN;            // 64
    float* centsqL = xiL + DN;               // 64
    float* red_val = centsqL + DN;           // 1024
    int*   red_idx = (int*)(red_val + MN);   // 1024
    int*   ctrlI   = red_idx + MN;           // [0]=n [1]=w [2]=create
    float* ctrlF   = (float*)(ctrlI + 4);    // [0]=sf [1]=si

    int tid = threadIdx.x;
    float* cents_ws = ws + OFF_CENT;
    int*   sup_ws   = (int*)(ws + OFF_SUP);
    float  gterm    = ws[OFF_GTERM];

    if (tid == 0) ctrlI[0] = 0;
    __syncthreads();

    for (int i = 0; i < BN; ++i) {
        if (tid < DN) xiL[tid] = x[i * DN + tid];
        if (tid == 0) ctrlF[1] = ws[OFF_SEN + i];
        __syncthreads();
        int nloc = ctrlI[0];
        float dv;
        if (tid < nloc) {
            float dist = 0.f;
            const float* pr = protosL + tid * 65;
            for (int d = 0; d < DN; ++d) { float df = xiL[d] - pr[d]; dist += df * df; }
            float stau = fabsf(gterm + sencL[tid] - cnormL[tid]) * 0.5f;
            dv = expf(-dist / stau);
        } else {
            dv = -INFINITY;
        }
        red_val[tid] = dv; red_idx[tid] = tid;
        __syncthreads();
        // argmax (first-index tie break, matches jnp.argmax)
        for (int off = 512; off > 0; off >>= 1) {
            if (tid < off) {
                float v2 = red_val[tid + off]; int i2 = red_idx[tid + off];
                float v1 = red_val[tid];       int i1 = red_idx[tid];
                if (v2 > v1 || (v2 == v1 && i2 < i1)) { red_val[tid] = v2; red_idx[tid] = i2; }
            }
            __syncthreads();
        }
        if (tid == 0) {
            float val = red_val[0]; int idx = red_idx[0];
            int create = (nloc == 0) || (val < DELTA_V);
            int w = create ? min(nloc, MN - 1) : idx;
            int supn = create ? 1 : (sup_ws[w] + 1);
            sup_ws[w] = supn;
            float sf = (float)supn;
            float si = ctrlF[1];
            float so = sencL[w];
            sencL[w] = create ? si : (so + (si - so) / sf);
            ctrlI[0] = create ? min(nloc + 1, MN) : nloc;
            ctrlI[1] = w; ctrlI[2] = create;
            ctrlF[0] = sf;
        }
        __syncthreads();
        int   w      = ctrlI[1];
        int   create = ctrlI[2];
        float sf     = ctrlF[0];
        if (tid < DN) {
            float xv = xiL[tid];
            float po = protosL[w * 65 + tid];
            float co = cents_ws[w * DN + tid];
            float pnew = create ? xv : po;
            float cnew = create ? xv : (co + (xv - co) / sf);
            protosL[w * 65 + tid] = pnew;
            cents_ws[w * DN + tid] = cnew;
            centsqL[tid] = cnew * cnew;
        }
        __syncthreads();
        if (tid == 0) {
            float cs = 0.f;
            for (int d = 0; d < DN; ++d) cs += centsqL[d];
            cnormL[w] = cs;
        }
        __syncthreads();
    }

    // persist final state for downstream kernels
    float* prot_ws = ws + OFF_PROT;
    for (int d = 0; d < DN; ++d) prot_ws[tid * DN + d] = protosL[tid * 65 + d];
    ws[OFF_SENC  + tid] = sencL[tid];
    ws[OFF_CNORM + tid] = cnormL[tid];
    if (tid == 0) ((int*)ws)[OFF_NRULES] = ctrlI[0];
}

// ---------------------------------------------------------------------------
// Kernel C: per-rule ||proto||^2 and stau
// ---------------------------------------------------------------------------
__global__ __launch_bounds__(256)
void k_rulestats(float* __restrict__ ws) {
    int r = blockIdx.x * blockDim.x + threadIdx.x;
    if (r >= MN) return;
    const float* pr = ws + OFF_PROT + r * DN;
    float s = 0.f;
    for (int d = 0; d < DN; ++d) s += pr[d] * pr[d];
    ws[OFF_PN + r] = s;
    ws[OFF_STAU + r] = fabsf(ws[OFF_GTERM] + ws[OFF_SENC + r] - ws[OFF_CNORM + r]) * 0.5f;
}

// ---------------------------------------------------------------------------
// Kernel D: dens[r,b] via WMMA f32 16x16x4. G = protos @ x^T (1024x1024x64).
// One 16x16 tile per wave; 8 waves per block; 512 blocks.
// A frag (16x4 f32): lanes 0-15 K0/K1, lanes 16-31 K2/K3 -> float2 loads.
// B frag (4x16):     same K split, N striped across lanes.
// ---------------------------------------------------------------------------
__global__ __launch_bounds__(256)
void k_dens_wmma(const float* __restrict__ x, float* __restrict__ ws) {
    int lane = threadIdx.x & 31;
    int wv   = threadIdx.x >> 5;
    int tile = blockIdx.x * 8 + wv;
    int m0 = (tile >> 6) * 16;
    int n0 = (tile & 63) * 16;
    int half = lane >> 4;
    int l15  = lane & 15;
    const float* A  = ws + OFF_PROT + (m0 + l15) * DN;
    const float* Bx = x + (n0 + l15) * DN;

    v8f c = {0.f, 0.f, 0.f, 0.f, 0.f, 0.f, 0.f, 0.f};
    for (int k = 0; k < DN; k += 4) {
        int ka = k + half * 2;
        v2f a = *(const v2f*)(A + ka);
        v2f b = *(const v2f*)(Bx + ka);
        c = __builtin_amdgcn_wmma_f32_16x16x4_f32(false, a, false, b,
                                                  (short)0, c, false, false);
    }
    int nr = ((const int*)ws)[OFF_NRULES];
    const float* sen  = ws + OFF_SEN;
    const float* pn   = ws + OFF_PN;
    const float* stau = ws + OFF_STAU;
    float* dens = ws + OFF_DENS;
    int bb = n0 + l15;
    float sb = sen[bb];
    for (int j = 0; j < 8; ++j) {
        int r = m0 + half * 8 + j;
        float d2 = sb + pn[r] - 2.0f * c[j];
        d2 = fmaxf(d2, 0.0f);
        float dvv = (r < nr) ? expf(-d2 / stau[r]) : 0.0f;
        dens[r * BN + bb] = dvv;     // regular-temporal: re-read twice, fits L2
    }
}

// ---------------------------------------------------------------------------
// Kernel D2: inv_colsum[b] = 1 / sum_r dens[r,b]  (coalesced over b)
// ---------------------------------------------------------------------------
__global__ __launch_bounds__(256)
void k_colsum(float* __restrict__ ws) {
    int b = blockIdx.x * blockDim.x + threadIdx.x;
    if (b >= BN) return;
    const float* dens = ws + OFF_DENS;
    float s = 0.f;
    for (int r = 0; r < MN; ++r) s += dens[r * BN + b];
    ws[OFF_INVC + b] = 1.0f / s;
}

// ---------------------------------------------------------------------------
// Kernel E: y_n = sigmoid(x @ W^T + b) via WMMA f32_16x16x32_f16.
// 1024x64x64; 256 tiles; 8 waves/block -> 32 blocks. 2 k-steps of K=32.
// ---------------------------------------------------------------------------
__global__ __launch_bounds__(256)
void k_yn_wmma(const float* __restrict__ x, const float* __restrict__ W,
               const float* __restrict__ bias, float* __restrict__ ws) {
    int lane = threadIdx.x & 31;
    int wv   = threadIdx.x >> 5;
    int tile = blockIdx.x * 8 + wv;           // 0..255
    int m0 = (tile >> 2) * 16;
    int n0 = (tile & 3) * 16;
    int half = lane >> 4;
    int l15  = lane & 15;
    const float* xr = x + (m0 + l15) * DN;
    int o = n0 + l15;
    const float* wr = W + o * DN;

    v8f c = {0.f, 0.f, 0.f, 0.f, 0.f, 0.f, 0.f, 0.f};
    for (int k0 = 0; k0 < DN; k0 += 32) {
        v16h a, bf;
        int ab = k0 + half * 16;              // A: K base per half-wave
        for (int j = 0; j < 8; ++j) {
            a[j]     = (_Float16)xr[ab + j];
            a[j + 8] = (_Float16)xr[ab + 8 + j];
        }
        int bbn = k0 + half * 16;             // B: 16 consecutive K per lane
        for (int j = 0; j < 16; ++j) bf[j] = (_Float16)wr[bbn + j];
        c = __builtin_amdgcn_wmma_f32_16x16x32_f16(false, a, false, bf,
                                                   (short)0, c, false, false);
    }
    float bo = bias[o];
    float* yn = ws + OFF_YN;
    for (int j = 0; j < 8; ++j) {
        int m = m0 + half * 8 + j;
        float v = c[j] + bo;
        yn[m * ON + o] = 1.0f / (1.0f + expf(-v));
    }
}

// ---------------------------------------------------------------------------
// Kernel F: out[b, r*O + o] = (dens[r,b]*inv_colsum[b]) * y_n[b,o]
// 64M floats = 256 MB streamed out -> THE bandwidth-bound kernel (~11 us floor
// at 23.3 TB/s). Each lane produces 4 consecutive outputs -> b128 stores,
// with non-temporal hint (output > L2, never re-read on device) so the
// reused dens/y_n working set stays cache-resident.
// ---------------------------------------------------------------------------
__global__ __launch_bounds__(256)
void k_out(const float* __restrict__ ws, float* __restrict__ out) {
    int base = (blockIdx.x * 256 + threadIdx.x) * 4;
    int bb  = base >> 16;           // / (M*O)
    int rem = base & 65535;
    int r = rem >> 6;
    int o = rem & 63;               // multiple of 4
    float lam = ws[OFF_DENS + r * BN + bb] * ws[OFF_INVC + bb];
    v4f y = *(const v4f*)(ws + OFF_YN + bb * ON + o);
    v4f res;
    res.x = lam * y.x;
    res.y = lam * y.y;
    res.z = lam * y.z;
    res.w = lam * y.w;
    __builtin_nontemporal_store(res, (v4f*)(out + base));
}

// ---------------------------------------------------------------------------
extern "C" void kernel_launch(void* const* d_in, const int* in_sizes, int n_in,
                              void* d_out, int out_size, void* d_ws, size_t ws_size,
                              hipStream_t stream) {
    const float* x = (const float*)d_in[0];   // [1024,64]
    const float* W = (const float*)d_in[1];   // [64,64]
    const float* b = (const float*)d_in[2];   // [64]
    float* ws  = (float*)d_ws;
    float* out = (float*)d_out;

    k_stats<<<1, 1024, 0, stream>>>(x, ws);

    // dynamic LDS for scan: protos(1024*65) + senc + cnorm + xi + centsq
    //                       + red_val + red_idx + ctrl  ~= 283 KB
    size_t scanLds = (size_t)(MN * 65 + MN + MN + DN + DN + MN) * sizeof(float)
                   + (size_t)MN * sizeof(int) + 8 * sizeof(int);
    k_scan<<<1, 1024, scanLds, stream>>>(x, ws);

    k_rulestats<<<MN / 256, 256, 0, stream>>>(ws);

    k_dens_wmma<<<(MN / 16) * (BN / 16) / 8, 256, 0, stream>>>(x, ws);   // 512 blocks

    k_colsum<<<BN / 256, 256, 0, stream>>>(ws);

    k_yn_wmma<<<(BN / 16) * (ON / 16) / 8, 256, 0, stream>>>(x, W, b, ws); // 32 blocks

    k_out<<<(BN * MN * ON / 4) / 256, 256, 0, stream>>>(ws, out);        // 65536 blocks
}
